// LlamaAttention_11699490914699
// MI455X (gfx1250) — matrix operand order repaired and checked by reading, not compiled, not run on hardware
//
#include <hip/hip_runtime.h>
#include <hip/hip_bf16.h>
#include <cstdint>

// ---------------------------------------------------------------------------
// Problem constants (from reference)
// ---------------------------------------------------------------------------
#define BB   2
#define SS   2048
#define DD   2048
#define HH   32
#define KVH  8
#define HD   64
#define MM   (BB * SS)          // 4096 token rows
#define NREP (HH / KVH)         // 4

typedef __attribute__((ext_vector_type(16))) __bf16 v16bf;
typedef __attribute__((ext_vector_type(8)))  __bf16 bf16x8;
typedef __attribute__((ext_vector_type(8)))  float  v8f;
typedef int v4i_vs __attribute__((__vector_size__(4 * sizeof(int))));

union AV { v16bf v; bf16x8 h[2]; };

__device__ __forceinline__ v8f wmma_bf16(v16bf a, v16bf b, v8f c) {
  // D = A(16x32 bf16) * B(32x16 bf16) + C(16x16 f32)
  return __builtin_amdgcn_wmma_f32_16x16x32_bf16(
      /*neg_a=*/false, a, /*neg_b=*/false, b,
      /*c_mod=*/(short)0, c, /*reuse_a=*/false, /*reuse_b=*/false);
}

// ---------------------------------------------------------------------------
// CDNA5 async copy global->LDS (ASYNCcnt-tracked, no VGPR transit)
// ---------------------------------------------------------------------------
__device__ __forceinline__ void async_load_b128(const __bf16* g, unsigned lds_off) {
#if __has_builtin(__builtin_amdgcn_global_load_async_to_lds_b128)
  typedef __attribute__((address_space(1))) v4i_vs* gp_t;  // global int4*
  typedef __attribute__((address_space(3))) v4i_vs* lp_t;  // shared int4*
  __builtin_amdgcn_global_load_async_to_lds_b128(
      (gp_t)(uintptr_t)g, (lp_t)(uintptr_t)lds_off, 0, 0);
#else
  asm volatile("global_load_async_to_lds_b128 %0, %1, off"
               :: "v"(lds_off), "v"(g) : "memory");
#endif
}

__device__ __forceinline__ void wait_async0() {
#if __has_builtin(__builtin_amdgcn_s_wait_asynccnt)
  __builtin_amdgcn_s_wait_asynccnt(0);
#else
  asm volatile("s_wait_asynccnt 0" ::: "memory");
#endif
}

// ---------------------------------------------------------------------------
// fp32 -> bf16 converter (staging into L2-resident bf16 buffers)
// ---------------------------------------------------------------------------
__global__ __launch_bounds__(256)
void cvt_f32_bf16(const float* __restrict__ src, __bf16* __restrict__ dst, int n) {
  int i = blockIdx.x * blockDim.x + threadIdx.x;
  int stride = gridDim.x * blockDim.x;
  for (; i < n; i += stride) dst[i] = (__bf16)src[i];
}

// ---------------------------------------------------------------------------
// NT GEMM: C[M,N] = A[M,K] * Bw[N,K]^T   (bf16 inputs, f32 WMMA accumulate)
// Block tile 128(M) x 64(N), 8 waves stacked along M; wave tile 16x64 (one
// full head). The shared 64x32 B-tile is double-buffered in LDS via
// global_load_async_to_lds_b128 (4KB = one b128 per thread per K-step).
// Epilogues: 0 = Q (RoPE + 1/sqrt(HD), store (B,H,S,HD) bf16)
//            1 = K (RoPE, store (B,KV,S,HD) bf16)
//            2 = V (store transposed (B,KV,HD,S) bf16)
//            3 = plain f32 store, row-major [M,Nld]
// ---------------------------------------------------------------------------
template <int EP>
__global__ __launch_bounds__(256)
void gemm_bf16(const __bf16* __restrict__ A,
               const __bf16* __restrict__ Bw,
               void* __restrict__ out,
               const int* __restrict__ pos,
               int K, int Nld) {
  __shared__ __align__(32) __bf16 Bt[2][64 * 32];   // double-buffered B tile

  const int tid  = threadIdx.x;
  const int lane = tid & 31;
  const int wave = tid >> 5;
  const int lm   = lane & 15;   // A: row m; B: col n; C: col n
  const int lh   = lane >> 4;   // half-wave selector

  const int n0 = blockIdx.x * 64;            // one full head strip (HD=64)
  const int m0 = blockIdx.y * 128 + wave * 16;

  // async staging assignment: thread -> (row n, 16B chunk) of the 64x32 tile
  const int brow = tid >> 2;                 // 0..63
  const int bcol = (tid & 3) * 8;            // element offset within row
  const __bf16* bsrc = Bw + (size_t)(n0 + brow) * K + bcol;
  const unsigned loff0 = (unsigned)(uintptr_t)&Bt[0][0] + (unsigned)(brow * 32 + bcol) * 2u;
  const unsigned loff1 = loff0 + 64 * 32 * 2;

  v8f acc[4] = {};
  const __bf16* arow = A + (size_t)(m0 + lm) * K;

  async_load_b128(bsrc, loff0);              // preload tile kk=0
  wait_async0();
  __syncthreads();

  for (int kk = 0; kk < K; kk += 32) {
    const int cur = (kk >> 5) & 1;
    const bool more = (kk + 32 < K);
    if (more) async_load_b128(bsrc + kk + 32, cur ? loff0 : loff1);

    AV a;
    a.h[0] = *(const bf16x8*)(arow + kk + lh * 16);       // K chunk [lh*16, +8)
    a.h[1] = *(const bf16x8*)(arow + kk + lh * 16 + 8);   // K chunk [lh*16+8, +8)
#pragma unroll
    for (int j = 0; j < 4; ++j) {
      const __bf16* bp = &Bt[cur][(j * 16 + lm) * 32 + lh * 16];
      v16bf bv = *(const v16bf*)bp;                       // 16 contiguous K vals
      acc[j] = wmma_bf16(a.v, bv, acc[j]);
    }
    if (more) wait_async0();
    __syncthreads();
  }

  // C fragment: element r of acc[j] = C[m0 + 8*lh + r, n0 + j*16 + lm]
  const int mb = m0 + 8 * lh;        // 8-row group never crosses batch bound
  const int b  = mb / SS;
  const int s0 = mb % SS;

  if constexpr (EP == 2) {           // ---- V: store transposed (B,KV,HD,S)
#pragma unroll
    for (int j = 0; j < 4; ++j) {
      int n = n0 + j * 16 + lm;
      int kvh = n / HD, d = n % HD;
      bf16x8 pk;
#pragma unroll
      for (int r = 0; r < 8; ++r) pk[r] = (__bf16)acc[j][r];
      __bf16* dst = (__bf16*)out + ((size_t)(b * KVH + kvh) * HD + d) * SS + s0;
      *(bf16x8*)dst = pk;            // 8 consecutive s positions, 16B store
    }
  } else if constexpr (EP == 0 || EP == 1) {  // ---- Q/K: fused RoPE
    const int   nh    = (EP == 0) ? HH : KVH;
    const float scale = (EP == 0) ? 0.125f : 1.0f;   // 1/sqrt(HD) on Q
    const int   head  = n0 / HD;
    __bf16* base = (__bf16*)out + ((size_t)(b * nh + head) * SS) * HD;
#pragma unroll
    for (int j = 0; j < 2; ++j) {
      // rotate-half pair: (acc[j], acc[j+2]) -> dims d and d+32, freq idx i=d
      int   i  = j * 16 + lm;
      float fr = __powf(10000.0f, (float)(-2 * i) * (1.0f / 64.0f));
#pragma unroll
      for (int r = 0; r < 8; ++r) {
        int   s = s0 + r;
        float p = (float)pos[b * SS + s];
        float sn, cs;
        __sincosf(p * fr, &sn, &cs);
        float x1 = acc[j][r], x2 = acc[j + 2][r];
        float o1 = (x1 * cs - x2 * sn) * scale;
        float o2 = (x2 * cs + x1 * sn) * scale;
        __bf16* row = base + (size_t)s * HD + i;
        row[0]  = (__bf16)o1;
        row[32] = (__bf16)o2;
      }
    }
  } else {                           // ---- plain f32 store [M, Nld]
    float* o = (float*)out;
#pragma unroll
    for (int j = 0; j < 4; ++j) {
      int n = n0 + j * 16 + lm;
#pragma unroll
      for (int r = 0; r < 8; ++r)
        o[(size_t)(mb + r) * Nld + n] = acc[j][r];
    }
  }
}

// ---------------------------------------------------------------------------
// Flash attention (causal, GQA). 1 wave = 16 query rows x full head (HD=64).
// Keys processed in blocks of 32 (one bf16 WMMA K-step for P*V).
// Qb: (B,H,S,HD) bf16 (pre-scaled), Kb: (B,KV,S,HD) bf16, Vt: (B,KV,HD,S) bf16
// Ab: (B,S,H*HD) bf16
// ---------------------------------------------------------------------------
__global__ __launch_bounds__(256)
void flash_attn(const __bf16* __restrict__ Qb,
                const __bf16* __restrict__ Kb,
                const __bf16* __restrict__ Vt,
                __bf16* __restrict__ Ab) {
  __shared__ __align__(32) __bf16 lp[8][16 * 32];  // per-wave P transpose tile

  const int lane = threadIdx.x & 31;
  const int wave = threadIdx.x >> 5;
  const int lm = lane & 15, lh = lane >> 4;

  const int qt  = blockIdx.x * 8 + wave;   // q tile index [0,128)
  const int h   = blockIdx.y;
  const int b   = blockIdx.z;
  const int kvh = h / NREP;

  const __bf16* qbase = Qb + (((size_t)(b * HH + h)) * SS + qt * 16) * HD;
  const __bf16* kbase = Kb + ((size_t)(b * KVH + kvh)) * SS * HD;
  const __bf16* vbase = Vt + ((size_t)(b * KVH + kvh)) * HD * SS;

  // Q A-fragments for both 32-wide contraction chunks of HD=64
  AV aq[2];
#pragma unroll
  for (int c = 0; c < 2; ++c) {
    const __bf16* qr = qbase + (size_t)lm * HD + c * 32 + lh * 16;
    aq[c].h[0] = *(const bf16x8*)(qr);
    aq[c].h[1] = *(const bf16x8*)(qr + 8);
  }

  v8f acco[4] = {};
  float mi[8], li[8];
#pragma unroll
  for (int r = 0; r < 8; ++r) { mi[r] = -3.0e38f; li[r] = 0.0f; }

  const int nkb = (qt >> 1) + 1;           // 32-key blocks covering [0, 16*(qt+1))
  for (int kb = 0; kb < nkb; ++kb) {
    const int key0 = kb * 32;

    // prefetch next key/value block into caches (global_prefetch_b8)
    if (kb + 1 < nkb) {
      __builtin_prefetch(kbase + (size_t)(key0 + 32 + lane) * HD, 0, 3);
      __builtin_prefetch(vbase + (size_t)(lane * 2) * SS + key0 + 32, 0, 3);
    }

    // scores: S[16q, 32k] as two 16x16 f32 fragments
    v8f sc[2] = {};
#pragma unroll
    for (int t = 0; t < 2; ++t)
#pragma unroll
      for (int c = 0; c < 2; ++c) {
        const __bf16* kp = kbase + (size_t)(key0 + t * 16 + lm) * HD + c * 32 + lh * 16;
        v16bf bk = *(const v16bf*)kp;
        sc[t] = wmma_bf16(aq[c].v, bk, sc[t]);
      }

    // causal mask + online softmax (row r lives in half-wave lh at VGPR r)
    float pscale[8];
#pragma unroll
    for (int r = 0; r < 8; ++r) {
      const int qrow = qt * 16 + 8 * lh + r;
#pragma unroll
      for (int t = 0; t < 2; ++t) {
        int key = key0 + t * 16 + lm;
        if (key > qrow) sc[t][r] = -3.0e38f;
      }
      float rm = fmaxf(sc[0][r], sc[1][r]);
      rm = fmaxf(rm, __shfl_xor(rm, 1));
      rm = fmaxf(rm, __shfl_xor(rm, 2));
      rm = fmaxf(rm, __shfl_xor(rm, 4));
      rm = fmaxf(rm, __shfl_xor(rm, 8));
      float mn  = fmaxf(mi[r], rm);
      float esc = __expf(mi[r] - mn);
      mi[r] = mn;
      float p0 = __expf(sc[0][r] - mn);
      float p1 = __expf(sc[1][r] - mn);
      sc[0][r] = p0; sc[1][r] = p1;
      float rs = p0 + p1;
      rs += __shfl_xor(rs, 1);
      rs += __shfl_xor(rs, 2);
      rs += __shfl_xor(rs, 4);
      rs += __shfl_xor(rs, 8);
      li[r] = li[r] * esc + rs;
      pscale[r] = esc;
    }
#pragma unroll
    for (int dsI = 0; dsI < 4; ++dsI)
#pragma unroll
      for (int r = 0; r < 8; ++r) acco[dsI][r] *= pscale[r];

    // transpose P (C-frag layout) -> A-frag layout via per-wave LDS tile
#pragma unroll
    for (int t = 0; t < 2; ++t)
#pragma unroll
      for (int r = 0; r < 8; ++r)
        lp[wave][(8 * lh + r) * 32 + t * 16 + lm] = (__bf16)sc[t][r];
    asm volatile("s_wait_dscnt 0" ::: "memory");
    AV ap;
    ap.h[0] = *(const bf16x8*)(&lp[wave][lm * 32 + lh * 16]);
    ap.h[1] = *(const bf16x8*)(&lp[wave][lm * 32 + lh * 16 + 8]);

    // O += P[16,32] * V[32,64]  (V transposed -> contiguous 32B B-operand loads)
#pragma unroll
    for (int dsI = 0; dsI < 4; ++dsI) {
      const __bf16* vp = vbase + (size_t)(dsI * 16 + lm) * SS + key0 + lh * 16;
      v16bf bv = *(const v16bf*)vp;
      acco[dsI] = wmma_bf16(ap.v, bv, acco[dsI]);
    }
  }

  // finalize: O / l, store bf16 to (B,S,H*HD) for output projection
  const size_t obase = ((size_t)b * SS + qt * 16 + 8 * lh) * (HH * HD) + (size_t)h * HD;
#pragma unroll
  for (int dsI = 0; dsI < 4; ++dsI) {
    int d = dsI * 16 + lm;
#pragma unroll
    for (int r = 0; r < 8; ++r) {
      float o = acco[dsI][r] / li[r];
      Ab[obase + (size_t)r * (HH * HD) + d] = (__bf16)o;
    }
  }
}

// ---------------------------------------------------------------------------
// Host-side orchestration
// ---------------------------------------------------------------------------
extern "C" void kernel_launch(void* const* d_in, const int* in_sizes, int n_in,
                              void* d_out, int out_size, void* d_ws, size_t ws_size,
                              hipStream_t stream) {
  const float* X  = (const float*)d_in[0];   // (B,S,D)
  const int*   P  = (const int*)d_in[1];     // (B,S)
  const float* Wq = (const float*)d_in[2];   // (H*HD, D)
  const float* Wk = (const float*)d_in[3];   // (KV*HD, D)
  const float* Wv = (const float*)d_in[4];   // (KV*HD, D)
  const float* Wo = (const float*)d_in[5];   // (D, H*HD)
  float* out = (float*)d_out;

  char* ws = (char*)d_ws;
  size_t off = 0;
  auto alloc = [&](size_t elems) {
    void* p = ws + off;
    off = (off + elems * sizeof(__bf16) + 255) & ~(size_t)255;
    return (__bf16*)p;
  };
  __bf16* Xb  = alloc((size_t)MM * DD);
  __bf16* Wqb = alloc((size_t)HH * HD * DD);
  __bf16* Wkb = alloc((size_t)KVH * HD * DD);
  __bf16* Wvb = alloc((size_t)KVH * HD * DD);
  __bf16* Wob = alloc((size_t)DD * HH * HD);
  __bf16* Qb  = alloc((size_t)BB * HH * SS * HD);
  __bf16* Kb  = alloc((size_t)BB * KVH * SS * HD);
  __bf16* Vtb = alloc((size_t)BB * KVH * HD * SS);
  __bf16* Ab  = alloc((size_t)MM * HH * HD);

  // stage everything as bf16 (L2-resident)
  cvt_f32_bf16<<<1024, 256, 0, stream>>>(X,  Xb,  MM * DD);
  cvt_f32_bf16<<<1024, 256, 0, stream>>>(Wq, Wqb, HH * HD * DD);
  cvt_f32_bf16<<<256,  256, 0, stream>>>(Wk, Wkb, KVH * HD * DD);
  cvt_f32_bf16<<<256,  256, 0, stream>>>(Wv, Wvb, KVH * HD * DD);
  cvt_f32_bf16<<<1024, 256, 0, stream>>>(Wo, Wob, DD * HH * HD);

  // QKV projections with fused RoPE / scaling / V-transpose
  gemm_bf16<0><<<dim3((HH * HD) / 64,  MM / 128), 256, 0, stream>>>(Xb, Wqb, Qb,  P, DD, HH * HD);
  gemm_bf16<1><<<dim3((KVH * HD) / 64, MM / 128), 256, 0, stream>>>(Xb, Wkb, Kb,  P, DD, KVH * HD);
  gemm_bf16<2><<<dim3((KVH * HD) / 64, MM / 128), 256, 0, stream>>>(Xb, Wvb, Vtb, P, DD, KVH * HD);

  // causal flash attention, 1 wave per 16-row q tile
  flash_attn<<<dim3((SS / 16) / 8, HH, BB), 256, 0, stream>>>(Qb, Kb, Vtb, Ab);

  // output projection -> fp32
  gemm_bf16<3><<<dim3(DD / 64, MM / 128), 256, 0, stream>>>(Ab, Wob, out, nullptr, HH * HD, DD);
}
